// GraphAttentionLayer_249108103457
// MI455X (gfx1250) — matrix operand order repaired and checked by reading, not compiled, not run on hardware
//
#include <hip/hip_runtime.h>
#include <hip/hip_bf16.h>

#define GB 8
#define GN 2048
#define GF 64
#define NEG_BIG_F (-9.0e15f)

// LDS padding for hT: TDM inserts 4 dwords (16B) after every 256 dwords (1024B).
// element e (bf16 halves, e = f*2048 + n) lives at half-offset e + 8*(e>>9).
// 16B alignment of 8-half chunks is preserved (pads are 16B, chunks never cross
// a 512-half boundary since e is always a multiple of 8 at chunk starts).
#define PAD_OFF(e) ((e) + 8 * ((e) >> 9))
#define LDS_HT_BYTES 266240   // 262144 + 16 * (262144/1024)

typedef __attribute__((ext_vector_type(16))) __bf16    v16bf;
typedef __attribute__((ext_vector_type(8)))  float     v8f;
typedef __attribute__((ext_vector_type(4)))  unsigned  u32x4;
typedef __attribute__((ext_vector_type(8)))  int       i32x8;
typedef __attribute__((ext_vector_type(4)))  int       i32x4;
typedef __attribute__((ext_vector_type(4)))  float     f32x4;

__device__ __forceinline__ unsigned pack_bf16(float lo, float hi) {
  unsigned a = __float_as_uint(lo);
  unsigned b = __float_as_uint(hi);
  a = (a + 0x7FFFu + ((a >> 16) & 1u)) >> 16;   // RNE
  b = (b + 0x7FFFu + ((b >> 16) & 1u)) >> 16;
  return (a & 0xFFFFu) | (b << 16);
}

// broadcast a per-row (rows 0..15 live in lanes 0..15) value to all lanes needing `row`
__device__ __forceinline__ float bcast_row(float v, int row) {
  return __int_as_float(__builtin_amdgcn_ds_bpermute(row << 2, __float_as_int(v)));
}

// ---------------- Kernel 1: h = input @ W ; e1 = h.a1 ; e2 = h.a2 ; hT (bf16, [b][f][n]) ----
__global__ __launch_bounds__(256) void gat_h_kernel(
    const float* __restrict__ input, const float* __restrict__ W,
    const float* __restrict__ a, unsigned short* __restrict__ hT,
    float* __restrict__ e1, float* __restrict__ e2)
{
  const int gid = blockIdx.x * 256 + threadIdx.x;   // flat row id: b*N + n
  const int b = gid >> 11;
  const int n = gid & (GN - 1);

  float inr[GF];
  const f32x4* ip = (const f32x4*)(input + (size_t)gid * GF);
  #pragma unroll
  for (int i = 0; i < GF / 4; ++i) ((f32x4*)inr)[i] = ip[i];

  float acc[GF];
  #pragma unroll
  for (int f = 0; f < GF; ++f) acc[f] = 0.f;

  for (int k = 0; k < GF; ++k) {                    // W[k*64+f] is wave-uniform -> scalar loads
    float x = inr[k];
    #pragma unroll
    for (int f = 0; f < GF; ++f) acc[f] = fmaf(x, W[k * GF + f], acc[f]);
  }

  float s1 = 0.f, s2 = 0.f;
  #pragma unroll
  for (int f = 0; f < GF; ++f) {
    s1 = fmaf(acc[f], a[f], s1);
    s2 = fmaf(acc[f], a[GF + f], s2);
  }
  e1[gid] = s1;
  e2[gid] = s2;

  unsigned short* hp = hT + (size_t)b * GF * GN + n;
  #pragma unroll
  for (int f = 0; f < GF; ++f) {
    unsigned u = __float_as_uint(acc[f]);
    u = (u + 0x7FFFu + ((u >> 16) & 1u)) >> 16;
    hp[(size_t)f * GN] = (unsigned short)u;         // coalesced across lanes (consecutive n)
  }
}

// ---------------- Kernel 2: fused masked-softmax(e) @ h  (flash-attention style) --------------
__global__ __launch_bounds__(256) void gat_attn_kernel(
    const int* __restrict__ adj, const unsigned short* __restrict__ hT,
    const float* __restrict__ e1g, const float* __restrict__ e2g,
    const float* __restrict__ bias, float* __restrict__ out)
{
  extern __shared__ char smem[];
  float*          lds_e2 = (float*)smem;                       // GN floats (8 KB)
  unsigned short* lds_h  = (unsigned short*)(smem + GN * 4);   // padded hT tile (260 KB)

  const int tid  = threadIdx.x;
  const int wid  = tid >> 5;
  const int lane = tid & 31;
  const int bIdx = blockIdx.x >> 4;     // 16 row-blocks of 128 rows per batch
  const int rb   = blockIdx.x & 15;

#if __has_builtin(__builtin_amdgcn_tensor_load_to_lds)
  // ---- Tensor Data Mover: DMA hT[b] (2048x64 bf16, padded) + e2[b] into LDS ----
  if (wid == 0) {
    unsigned long long gah = (unsigned long long)(hT + (size_t)bIdx * GF * GN);
    unsigned long long gae = (unsigned long long)(e2g + (size_t)bIdx * GN);

    // D# group0: {count=1, lds_addr, global_addr[56:0], type=2}
    u32x4 g0h = { 1u, (unsigned)(size_t)(void*)lds_h,
                  (unsigned)gah, (unsigned)((gah >> 32) & 0x01FFFFFFu) | 0x80000000u };
    u32x4 g0e = { 1u, (unsigned)(size_t)(void*)lds_e2,
                  (unsigned)gae, (unsigned)((gae >> 32) & 0x01FFFFFFu) | 0x80000000u };

    // D# group1 for hT: data_size=2B(code1), pad_enable=1, pad_interval=7(256dw),
    // pad_amount=3(4dw); tensor_dim0=2048, tensor_dim1=64, tile=2048x64, stride0=2048
    i32x8 g1h = { (int)((1u << 16) | (1u << 20) | (7u << 22) | (3u << 25)),
                  (int)(2048u << 16),           // tensor_dim0[15:0]
                  (int)(64u << 16),             // tensor_dim0[31:16]=0 | tensor_dim1[15:0]
                  (int)(2048u << 16),           // tensor_dim1[31:16]=0 | tile_dim0
                  64,                           // tile_dim1 | tile_dim2=0
                  2048, 0, 0 };                 // tensor_dim0_stride

    // D# group1 for e2: data_size=4B(code2), no pad; 2048x1 tile
    i32x8 g1e = { (int)(2u << 16),
                  (int)(2048u << 16),
                  (int)(1u << 16),
                  (int)(2048u << 16),
                  1,
                  2048, 0, 0 };

    i32x4 z4 = { 0, 0, 0, 0 };                  // groups 2/3: zero (<=2D tensor)
    i32x8 z8 = { 0, 0, 0, 0, 0, 0, 0, 0 };      // extra descriptor words (unused)
    __builtin_amdgcn_tensor_load_to_lds(g0h, g1h, z4, z4, z8, 0);
    __builtin_amdgcn_tensor_load_to_lds(g0e, g1e, z4, z4, z8, 0);
#if __has_builtin(__builtin_amdgcn_s_wait_tensorcnt)
    __builtin_amdgcn_s_wait_tensorcnt(0);
#else
    asm volatile("s_wait_tensorcnt 0x0" ::: "memory");
#endif
  }
#else
  // ---- fallback: manual staging into the same padded layout ----
  {
    const f32x4* s = (const f32x4*)(e2g + (size_t)bIdx * GN);
    f32x4* d = (f32x4*)lds_e2;
    for (int i = tid; i < GN / 4; i += 256) d[i] = s[i];
    const u32x4* sh = (const u32x4*)(hT + (size_t)bIdx * GF * GN);
    for (int i = tid; i < GF * GN / 8; i += 256) {
      int e = i * 8;
      *(u32x4*)(lds_h + PAD_OFF(e)) = sh[i];
    }
  }
#endif
  __syncthreads();

  // A-fragment lane mapping (16x32 bf16): lane holds row r=lane&15; VGPR g holds K pair
  //   kbase(g) = 2*(g&3) + (g>>2)*16 + (lane>=16 ? 8 : 0)
  const int r    = lane & 15;
  const int hi8  = (lane >> 4) << 3;
  const int grow = rb * 128 + wid * 16 + r;
  const float e1r = e1g[bIdx * GN + grow];
  const int* adjRow = adj + ((size_t)bIdx * GN + grow) * GN;

  // B-fragment lane mapping (32x16 bf16): lane = feature col N=lane&15, K = kB+{0..7,16..23}
  const int fB = lane & 15;
  const int kB = (lane >> 4) << 3;

  float m = -3.0e38f, lsum = 0.f;
  v8f acc[4];
  #pragma unroll
  for (int t = 0; t < 4; ++t)
    #pragma unroll
    for (int g = 0; g < 8; ++g) acc[t][g] = 0.f;

  for (int j0 = 0; j0 < GN; j0 += 32) {
    // --- adj tile + e2 in A-fragment order: cols j0+hi8+{0..7,16..23} ---
    const int* ap = adjRow + j0 + hi8;
    i32x4 A0 = *(const i32x4*)(ap + 0);
    i32x4 A1 = *(const i32x4*)(ap + 4);
    i32x4 A2 = *(const i32x4*)(ap + 16);
    i32x4 A3 = *(const i32x4*)(ap + 20);
    const float* ep = lds_e2 + j0 + hi8;
    f32x4 C0 = *(const f32x4*)(ep + 0);
    f32x4 C1 = *(const f32x4*)(ep + 4);
    f32x4 C2 = *(const f32x4*)(ep + 16);
    f32x4 C3 = *(const f32x4*)(ep + 20);

    int av[16]   = {A0.x, A0.y, A0.z, A0.w, A1.x, A1.y, A1.z, A1.w,
                    A2.x, A2.y, A2.z, A2.w, A3.x, A3.y, A3.z, A3.w};
    float cv[16] = {C0.x, C0.y, C0.z, C0.w, C1.x, C1.y, C1.z, C1.w,
                    C2.x, C2.y, C2.z, C2.w, C3.x, C3.y, C3.z, C3.w};

    float ev[16];
    float lm = -3.0e38f;
    #pragma unroll
    for (int t = 0; t < 16; ++t) {
      float x = e1r + cv[t];
      x = fmaxf(x, 0.1f * x);                 // leaky_relu(0.1)
      x = (av[t] > 0) ? x : NEG_BIG_F;        // adjacency mask
      ev[t] = x;
      lm = fmaxf(lm, x);
    }
    // combine the two half-rows (lane and lane^16 hold the same row)
    lm = fmaxf(lm, __shfl_xor(lm, 16, 32));
    float mnew  = fmaxf(m, lm);
    float scale = __expf(m - mnew);
    m = mnew;

    float ps = 0.f;
    unsigned pk[8];
    #pragma unroll
    for (int g = 0; g < 8; ++g) {
      float p0 = __expf(ev[2 * g]     - mnew);
      float p1 = __expf(ev[2 * g + 1] - mnew);
      ps += p0 + p1;
      pk[g] = pack_bf16(p0, p1);
    }
    ps += __shfl_xor(ps, 16, 32);
    lsum = lsum * scale + ps;

    // rescale f32 accumulators: C-layout row for VGPR g is g+hi8; get that row's scale
    #pragma unroll
    for (int g = 0; g < 8; ++g) {
      float sg = bcast_row(scale, g + hi8);
      acc[0][g] *= sg; acc[1][g] *= sg; acc[2][g] *= sg; acc[3][g] *= sg;
    }

    // A fragment (probabilities, bf16)
    union AF { unsigned u[8]; v16bf v; } af;
    #pragma unroll
    for (int g = 0; g < 8; ++g) af.u[g] = pk[g];

    // 4 x WMMA over the 64-wide feature dim; B frags from padded LDS (2 x b128 per lane)
    #pragma unroll
    for (int t = 0; t < 4; ++t) {
      union BF { u32x4 q[2]; v16bf v; } bf;
      int e = (t * 16 + fB) * GN + j0 + kB;
      const unsigned short* p = lds_h + PAD_OFF(e);
      bf.q[0] = *(const u32x4*)(p);
      bf.q[1] = *(const u32x4*)(lds_h + PAD_OFF(e + 16));
      acc[t] = __builtin_amdgcn_wmma_f32_16x16x32_bf16(
          false, af.v, false, bf.v, (short)0, acc[t], false, false);
    }
  }

  // epilogue: divide by softmax denom (v_rcp_f32), + bias, ELU, store
  float linv[8];
  #pragma unroll
  for (int g = 0; g < 8; ++g) linv[g] = __builtin_amdgcn_rcpf(bcast_row(lsum, g + hi8));

  #pragma unroll
  for (int t = 0; t < 4; ++t) {
    float bv = bias[t * 16 + fB];
    #pragma unroll
    for (int g = 0; g < 8; ++g) {
      int row = rb * 128 + wid * 16 + g + hi8;
      float x = acc[t][g] * linv[g] + bv;
      x = (x > 0.f) ? x : (__expf(x) - 1.0f);     // elu, alpha=1
      out[((size_t)bIdx * GN + row) * GF + t * 16 + fB] = x;
    }
  }
}

extern "C" void kernel_launch(void* const* d_in, const int* in_sizes, int n_in,
                              void* d_out, int out_size, void* d_ws, size_t ws_size,
                              hipStream_t stream) {
  (void)in_sizes; (void)n_in; (void)out_size; (void)ws_size;
  const float* input = (const float*)d_in[0];
  const int*   adj   = (const int*)d_in[1];
  const float* W     = (const float*)d_in[2];
  const float* a     = (const float*)d_in[3];
  const float* bias  = (const float*)d_in[4];
  float* out = (float*)d_out;

  // workspace: hT (bf16, 2 MB) | e1 (64 KB) | e2 (64 KB)
  unsigned short* hT = (unsigned short*)d_ws;
  float* e1 = (float*)((char*)d_ws + (size_t)GB * GF * GN * 2);
  float* e2 = e1 + GB * GN;

  gat_h_kernel<<<GB * GN / 256, 256, 0, stream>>>(input, W, a, hT, e1, e2);

  size_t smem = (size_t)GN * 4 + LDS_HT_BYTES;   // 274432 B < 320 KB WGP LDS
  (void)hipFuncSetAttribute((const void*)gat_attn_kernel,
                            hipFuncAttributeMaxDynamicSharedMemorySize, (int)smem);
  gat_attn_kernel<<<GB * 16, 256, smem, stream>>>(adj, hT, e1, e2, bias, out);
}
